// CrossAttention_13159779795222
// MI455X (gfx1250) — matrix operand order repaired and checked by reading, not compiled, not run on hardware
//
#include <hip/hip_runtime.h>

typedef __bf16 bf16;
typedef __bf16 bf16x8 __attribute__((ext_vector_type(8)));
typedef __bf16 v16bf  __attribute__((ext_vector_type(16)));
typedef float  v8f    __attribute__((ext_vector_type(8)));
typedef unsigned int v4u __attribute__((ext_vector_type(4)));
typedef int v8i __attribute__((ext_vector_type(8)));
typedef int v4i __attribute__((ext_vector_type(4)));

#define HEADS 8
#define DHEAD 64

#if defined(__HIP_DEVICE_COMPILE__) && __has_builtin(__builtin_amdgcn_tensor_load_to_lds)
#define HAVE_TDM 1
#else
#define HAVE_TDM 0
#endif

// ---------------------------------------------------------------------------
// WMMA helper (CDNA5 wave32, v_wmma_f32_16x16x32_bf16)
// ---------------------------------------------------------------------------
static __device__ __forceinline__ v8f wmma_bf16(v16bf a, v16bf b, v8f c) {
  return __builtin_amdgcn_wmma_f32_16x16x32_bf16(false, a, false, b, (short)0, c,
                                                 false, false);
}

// A-matrix 16x32 bf16 fragment (ISA 7.12.2):
// lane l: row m = l%16; elems 0..7  -> K = k0 +      (l/16)*8 + j
//                       elems 8..15 -> K = k0 + 16 + (l/16)*8 + j
static __device__ __forceinline__ v16bf load_a_frag(const bf16* base, int stride,
                                                    int m, int hi, int k0) {
  union { v16bf v; bf16x8 h[2]; } u;
  const bf16* row = base + (size_t)m * stride + k0 + hi * 8;
  u.h[0] = *(const bf16x8*)(row);
  u.h[1] = *(const bf16x8*)(row + 16);
  return u.v;
}

// B-matrix 32x16 bf16 fragment: lane l: col n = l%16;
// elems 0..7 -> K = k0 + (l/16)*8 + j; elems 8..15 -> K = k0 + 16 + (l/16)*8 + j
static __device__ __forceinline__ v16bf load_b_frag(const bf16* base, int stride,
                                                    int n, int hi, int k0) {
  union { v16bf v; bf16x8 h[2]; } u;
  const bf16* row = base + (size_t)n * stride + k0 + hi * 8;
  u.h[0] = *(const bf16x8*)(row);
  u.h[1] = *(const bf16x8*)(row + 16);
  return u.v;
}

// ---------------------------------------------------------------------------
// VALU cross-lane reductions over 16-lane rows (DPP8 + v_permlane16_b32)
// ---------------------------------------------------------------------------
#define DPP8_SEL(a,b,c,d,e,f,g,h) \
  ((a)|((b)<<3)|((c)<<6)|((d)<<9)|((e)<<12)|((f)<<15)|((g)<<18)|((h)<<21))
#define SEL_X1 DPP8_SEL(1,0,3,2,5,4,7,6)
#define SEL_X2 DPP8_SEL(2,3,0,1,6,7,4,5)
#define SEL_X4 DPP8_SEL(4,5,6,7,0,1,2,3)
#define PL16_LO ((int)0xFEDCBA98)
#define PL16_HI ((int)0x76543210)

static __device__ __forceinline__ float xor8_pl16(float v) {
  int i = __float_as_int(v);
  return __int_as_float(
      __builtin_amdgcn_permlane16(i, i, PL16_LO, PL16_HI, false, false));
}
static __device__ __forceinline__ float row16_max(float v) {
  v = fmaxf(v, __int_as_float(__builtin_amdgcn_mov_dpp8(__float_as_int(v), SEL_X1)));
  v = fmaxf(v, __int_as_float(__builtin_amdgcn_mov_dpp8(__float_as_int(v), SEL_X2)));
  v = fmaxf(v, __int_as_float(__builtin_amdgcn_mov_dpp8(__float_as_int(v), SEL_X4)));
  v = fmaxf(v, xor8_pl16(v));
  return v;
}
static __device__ __forceinline__ float row16_sum(float v) {
  v += __int_as_float(__builtin_amdgcn_mov_dpp8(__float_as_int(v), SEL_X1));
  v += __int_as_float(__builtin_amdgcn_mov_dpp8(__float_as_int(v), SEL_X2));
  v += __int_as_float(__builtin_amdgcn_mov_dpp8(__float_as_int(v), SEL_X4));
  v += xor8_pl16(v);
  return v;
}

#if HAVE_TDM
// ---------------------------------------------------------------------------
// Tensor Data Mover: 2D tile (d1 rows x d0 elems of 2B), row stride in elems.
// D# per ISA ch.8: group0 = count/lds_addr/global_addr/type=2,
// group1 = data_size=2B, tensor dims == tile dims, dim0 stride.
// This toolchain exposes the 6-arg builtin:
//   (uint32x4 g0, int32x8 g1, int32x4, int32x4, int32x8, i32 cpol)
// ---------------------------------------------------------------------------
static __device__ __forceinline__ void tdm_load_2d(const void* lds_dst,
                                                   const void* gaddr,
                                                   unsigned d0, unsigned d1,
                                                   unsigned long long stride) {
  unsigned long long ga = (unsigned long long)(uintptr_t)gaddr;
  unsigned lds_addr = (unsigned)(uintptr_t)lds_dst;  // low 32 bits = LDS offset
  v4u g0;
  g0[0] = 1u;                                           // count=1, user mode
  g0[1] = lds_addr;                                     // lds_addr
  g0[2] = (unsigned)ga;                                 // global_addr[31:0]
  g0[3] = (unsigned)((ga >> 32) & 0x01FFFFFFu) | (2u << 30);  // [56:32] | type=2
  v8i g1;
  g1[0] = (int)(1u << 16);                              // data_size=1 (2 bytes)
  g1[1] = (int)((d0 & 0xFFFFu) << 16);                  // tensor_dim0 lo16
  g1[2] = (int)((d0 >> 16) | ((d1 & 0xFFFFu) << 16));   // dim0 hi | dim1 lo
  g1[3] = (int)((d1 >> 16) | ((d0 & 0xFFFFu) << 16));   // dim1 hi | tile_dim0
  g1[4] = (int)(d1 & 0xFFFFu);                          // tile_dim1 | tile_dim2=0
  g1[5] = (int)(unsigned)(stride & 0xFFFFFFFFull);      // dim0_stride lo
  g1[6] = (int)(unsigned)(stride >> 32);                // dim0_stride hi
  g1[7] = 0;
  v4i g2 = {0, 0, 0, 0};
  v4i g3 = {0, 0, 0, 0};
  v8i g4 = {0, 0, 0, 0, 0, 0, 0, 0};
  __builtin_amdgcn_tensor_load_to_lds(g0, g1, g2, g3, g4, 0);
}
#endif

// ---------------------------------------------------------------------------
// fp32 -> bf16 conversion (scale folded in for Wq*SCALE)
// ---------------------------------------------------------------------------
__global__ void cvt_bf16_kernel(const float* __restrict__ s, bf16* __restrict__ d,
                                long n, float scale) {
  long i = (long)blockIdx.x * blockDim.x + threadIdx.x;
  long stride = (long)gridDim.x * blockDim.x;
  for (; i < n; i += stride) d[i] = (bf16)(s[i] * scale);
}

// ---------------------------------------------------------------------------
// GEMM: Y[r,o] = sum_c X[r,c] * W[o,c]   (torch Linear, y = x @ W.T)
// 256 threads = 8 waves; wave tile 16 rows x 64 cols; block 128 x 64.
// MODE 0: bf16 head-split [B,H,rowsPerB,64]  (Q, K)
// MODE 1: bf16 head-split transposed [B,H,64,rowsPerB]  (V^T)
// MODE 2: fp32 row-major + bias  (final projection)
// ---------------------------------------------------------------------------
template <int MODE>
__global__ void gemm_wmma_kernel(const bf16* __restrict__ X, const bf16* __restrict__ W,
                                 int C, int O, int rowsPerB,
                                 bf16* __restrict__ outb, float* __restrict__ outf,
                                 const float* __restrict__ bias) {
  const int lane = threadIdx.x & 31;
  const int wave = threadIdx.x >> 5;
  const int m  = lane & 15;
  const int hi = lane >> 4;
  const int rowBase = blockIdx.x * 128 + wave * 16;
  const int colBase = blockIdx.y * 64;

  v8f acc[4] = {};
  const bf16* xBase = X + (size_t)rowBase * C;

  for (int k0 = 0; k0 < C; k0 += 32) {
    v16bf a = load_a_frag(xBase, C, m, hi, k0);
#pragma unroll
    for (int t = 0; t < 4; ++t) {
      v16bf b = load_b_frag(W + (size_t)(colBase + t * 16) * C, C, m, hi, k0);
      acc[t] = wmma_bf16(a, b, acc[t]);
    }
  }

#pragma unroll
  for (int t = 0; t < 4; ++t) {
    const int o = colBase + t * 16 + m;
#pragma unroll
    for (int r = 0; r < 8; ++r) {
      const int row = rowBase + r + 8 * hi;
      const float v = acc[t][r];
      if (MODE == 2) {
        outf[(size_t)row * O + o] = v + bias[o];
      } else {
        const int bb = row / rowsPerB;
        const int n  = row % rowsPerB;
        const int h  = o >> 6;
        const int dd = o & 63;
        if (MODE == 0)
          outb[(((size_t)bb * HEADS + h) * rowsPerB + n) * DHEAD + dd] = (bf16)v;
        else
          outb[(((size_t)bb * HEADS + h) * DHEAD + dd) * rowsPerB + n] = (bf16)v;
      }
    }
  }
}

// ---------------------------------------------------------------------------
// Fused flash attention. Block = 8 waves sharing one (b,h); wave = 16 queries.
// K/V^T blocks (64 keys) staged to LDS by the Tensor Data Mover (double
// buffered, s_wait_tensorcnt) or a cooperative-copy fallback.
// q: [B,H,N,64] bf16 (scale folded into Wq), k: [B,H,M,64], vT: [B,H,64,M]
// o: [B,N,H*64] bf16
// ---------------------------------------------------------------------------
__global__ void attn_kernel(const bf16* __restrict__ q, const bf16* __restrict__ k,
                            const bf16* __restrict__ vT, bf16* __restrict__ o,
                            int N, int M) {
  __shared__ __align__(16) bf16 kbuf[2][64 * 64];   // 2 x 8KB
  __shared__ __align__(16) bf16 vbuf[2][64 * 64];   // 2 x 8KB
  __shared__ __align__(16) bf16 plds[8][16 * 64];   // 2KB per wave

  const int lane = threadIdx.x & 31;
  const int wave = threadIdx.x >> 5;
  const int m  = lane & 15;
  const int hi = lane >> 4;

  const int blocksPerBH = N >> 7;                  // 128 queries per block
  const int bh = blockIdx.x / blocksPerBH;
  const int qtile = (blockIdx.x % blocksPerBH) * 8 + wave;

  const bf16* qp = q  + ((size_t)bh * N + qtile * 16) * DHEAD;
  const bf16* kp = k  + (size_t)bh * M * DHEAD;
  const bf16* vp = vT + (size_t)bh * DHEAD * M;
  bf16* wlds = plds[wave];

  const v16bf qf0 = load_a_frag(qp, DHEAD, m, hi, 0);
  const v16bf qf1 = load_a_frag(qp, DHEAD, m, hi, 32);

  v8f acc[4] = {};
  float mrun[8], lsum[8];
#pragma unroll
  for (int r = 0; r < 8; ++r) { mrun[r] = -3.0e38f; lsum[r] = 0.0f; }

  auto compute_block = [&](const bf16* kb, const bf16* vb) {
    // ---- S = Q @ K^T for 64 keys (LDS-resident K block, row stride 64) ----
    v8f sc[4];
#pragma unroll
    for (int t = 0; t < 4; ++t) {
      const bf16* kt = kb + (size_t)(t * 16) * 64;
      v16bf b0 = load_b_frag(kt, 64, m, hi, 0);
      v16bf b1 = load_b_frag(kt, 64, m, hi, 32);
      v8f s = {};
      s = wmma_bf16(qf0, b0, s);
      s = wmma_bf16(qf1, b1, s);
      sc[t] = s;
    }

    // ---- online softmax (rows live in 16-lane halves of the C layout) ----
#pragma unroll
    for (int r = 0; r < 8; ++r) {
      float rmax = fmaxf(fmaxf(sc[0][r], sc[1][r]), fmaxf(sc[2][r], sc[3][r]));
      rmax = row16_max(rmax);
      const float mnew = fmaxf(mrun[r], rmax);
      const float rescale = __expf(mrun[r] - mnew);
      mrun[r] = mnew;
      const int row = r + 8 * hi;
      float psum = 0.0f;
#pragma unroll
      for (int t = 0; t < 4; ++t) {
        const float pv = __expf(sc[t][r] - mnew);
        psum += pv;
        wlds[row * 64 + t * 16 + m] = (bf16)pv;
      }
      lsum[r] = lsum[r] * rescale + psum;  // per-lane partial; reduce at end
#pragma unroll
      for (int t = 0; t < 4; ++t) acc[t][r] *= rescale;
    }
    __builtin_amdgcn_wave_barrier();
    asm volatile("s_wait_dscnt 0x0" ::: "memory");  // LDS in-order per wave

    const v16bf pf0 = load_a_frag(wlds, 64, m, hi, 0);
    const v16bf pf1 = load_a_frag(wlds, 64, m, hi, 32);
    __builtin_amdgcn_wave_barrier();

    // ---- acc += P @ V (LDS-resident V^T block, row stride 64) ----
#pragma unroll
    for (int t = 0; t < 4; ++t) {
      const bf16* vt = vb + (size_t)(t * 16) * 64;
      v16bf bv0 = load_b_frag(vt, 64, m, hi, 0);
      v16bf bv1 = load_b_frag(vt, 64, m, hi, 32);
      acc[t] = wmma_bf16(pf0, bv0, acc[t]);
      acc[t] = wmma_bf16(pf1, bv1, acc[t]);
    }
  };

#if HAVE_TDM
  // --- TDM double-buffered staging: wave 0 drives the DMA ---
  if (wave == 0) {
    tdm_load_2d(kbuf[0], kp, 64, 64, 64ull);              // K rows contiguous
    tdm_load_2d(vbuf[0], vp, 64, 64, (unsigned long long)M);  // V^T rows stride M
  }
  for (int mb = 0; mb < M; mb += 64) {
    const int p = (mb >> 6) & 1;
    if (wave == 0) {
      if (mb + 64 < M) {
        tdm_load_2d(kbuf[p ^ 1], kp + (size_t)(mb + 64) * 64, 64, 64, 64ull);
        tdm_load_2d(vbuf[p ^ 1], vp + (mb + 64), 64, 64, (unsigned long long)M);
        __builtin_amdgcn_s_wait_tensorcnt(2);  // current block's 2 loads done
      } else {
        __builtin_amdgcn_s_wait_tensorcnt(0);
      }
    }
    __syncthreads();           // data ready for everyone
    compute_block(kbuf[p], vbuf[p]);
    __syncthreads();           // safe to overwrite buf[p] two iters later
  }
#else
  // --- fallback: cooperative global->LDS copy, single buffer ---
  const int tid = (int)threadIdx.x;
  for (int mb = 0; mb < M; mb += 64) {
    {  // K block: 4096 contiguous bf16
      const bf16x8* src = (const bf16x8*)(kp + (size_t)mb * 64);
      bf16x8* dst = (bf16x8*)kbuf[0];
      for (int i = tid; i < 512; i += 256) dst[i] = src[i];
    }
    {  // V^T block: 64 rows, stride M
      const int row = tid >> 2, c = (tid & 3) * 16;
      const bf16* s = vp + (size_t)row * M + mb + c;
      bf16* d = vbuf[0] + row * 64 + c;
      *(bf16x8*)d = *(const bf16x8*)s;
      *(bf16x8*)(d + 8) = *(const bf16x8*)(s + 8);
    }
    __syncthreads();
    compute_block(kbuf[0], vbuf[0]);
    __syncthreads();
  }
#endif

  // ---- finalize: reduce per-lane partial sums once, normalize, store ----
  const int bb = bh >> 3;
  const int h  = bh & 7;
#pragma unroll
  for (int r = 0; r < 8; ++r) {
    const float inv = __builtin_amdgcn_rcpf(row16_sum(lsum[r]));
    const int n = qtile * 16 + r + 8 * hi;
#pragma unroll
    for (int t = 0; t < 4; ++t) {
      const size_t idx =
          ((size_t)bb * N + n) * (HEADS * DHEAD) + h * DHEAD + t * 16 + m;
      o[idx] = (bf16)(acc[t][r] * inv);
    }
  }
}

// ---------------------------------------------------------------------------
// Host launcher
// ---------------------------------------------------------------------------
extern "C" void kernel_launch(void* const* d_in, const int* in_sizes, int n_in,
                              void* d_out, int out_size, void* d_ws, size_t ws_size,
                              hipStream_t stream) {
  const float* x   = (const float*)d_in[0];
  const float* ctx = (const float*)d_in[1];
  const float* Wq  = (const float*)d_in[2];
  const float* Wk  = (const float*)d_in[3];
  const float* Wv  = (const float*)d_in[4];
  const float* Wo  = (const float*)d_in[5];
  const float* bo  = (const float*)d_in[6];
  float* out = (float*)d_out;

  const int B = 4;
  const int QD = 512, CD = 768, ID = HEADS * DHEAD;  // 512
  const int N = in_sizes[0] / (B * QD);   // 2048
  const int M = in_sizes[1] / (B * CD);   // 1024
  const float SCALE = 0.125f;             // 64^-0.5

  char* p = (char*)d_ws;
  auto carve = [&](size_t bytes) {
    char* r = p;
    p += (bytes + 255) & ~(size_t)255;
    return r;
  };
  bf16* xb  = (bf16*)carve((size_t)B * N * QD * 2);
  bf16* cb  = (bf16*)carve((size_t)B * M * CD * 2);
  bf16* wqb = (bf16*)carve((size_t)ID * QD * 2);
  bf16* wkb = (bf16*)carve((size_t)ID * CD * 2);
  bf16* wvb = (bf16*)carve((size_t)ID * CD * 2);
  bf16* wob = (bf16*)carve((size_t)QD * ID * 2);
  bf16* qws = (bf16*)carve((size_t)B * HEADS * N * DHEAD * 2);
  bf16* kws = (bf16*)carve((size_t)B * HEADS * M * DHEAD * 2);
  bf16* vtw = (bf16*)carve((size_t)B * HEADS * DHEAD * M * 2);
  bf16* ows = (bf16*)carve((size_t)B * N * ID * 2);
  (void)ws_size; (void)n_in; (void)out_size;

  const int CVB = 1024, CVT = 256;
  cvt_bf16_kernel<<<CVB, CVT, 0, stream>>>(x,   xb,  (long)B * N * QD, 1.0f);
  cvt_bf16_kernel<<<CVB, CVT, 0, stream>>>(ctx, cb,  (long)B * M * CD, 1.0f);
  cvt_bf16_kernel<<<CVB, CVT, 0, stream>>>(Wq,  wqb, (long)ID * QD, SCALE);
  cvt_bf16_kernel<<<CVB, CVT, 0, stream>>>(Wk,  wkb, (long)ID * CD, 1.0f);
  cvt_bf16_kernel<<<CVB, CVT, 0, stream>>>(Wv,  wvb, (long)ID * CD, 1.0f);
  cvt_bf16_kernel<<<CVB, CVT, 0, stream>>>(Wo,  wob, (long)QD * ID, 1.0f);

  dim3 gq((B * N) / 128, ID / 64);
  gemm_wmma_kernel<0><<<gq, 256, 0, stream>>>(xb, wqb, QD, ID, N, qws, nullptr, nullptr);
  dim3 gk((B * M) / 128, ID / 64);
  gemm_wmma_kernel<0><<<gk, 256, 0, stream>>>(cb, wkb, CD, ID, M, kws, nullptr, nullptr);
  gemm_wmma_kernel<1><<<gk, 256, 0, stream>>>(cb, wvb, CD, ID, M, vtw, nullptr, nullptr);

  // fused flash attention: 8 waves/block share one (b,h); 128 queries/block
  attn_kernel<<<B * HEADS * (N / 128), 256, 0, stream>>>(qws, kws, vtw, ows, N, M);

  dim3 go((B * N) / 128, QD / 64);
  gemm_wmma_kernel<2><<<go, 256, 0, stream>>>(ows, wob, ID, QD, N, nullptr, out, bo);
}
